// TreeLSTMDoubleCell_25254407701046
// MI455X (gfx1250) — compile-verified, matrix operand order checked by reading, not run on hardware
//
#include <hip/hip_runtime.h>

typedef __attribute__((ext_vector_type(16))) _Float16 v16h;
typedef __attribute__((ext_vector_type(8)))  float    v8f;

#define HD 128
#define FD 512

union AFrag { v16h v; _Float16 e[16]; };
union CFrag { v8f  v; float     e[8];  };

__device__ __forceinline__ float sigf(float x) { return 1.0f / (1.0f + __expf(-x)); }

// ---------------------------------------------------------------------------
// Pack a [128,512] f32 weight into f16, tiled per WMMA B-fragment:
// block b = ct*4 + kt (ct: 16-col tile, kt: 32-K tile), 512 halves per block.
// Within a block: lane l holds column N = l&15, K-pattern mirroring the A
// fragment layout (ISA 7.12.2): K = kt*32 + (h>>3)*16 + ((l>>4)*8) + (h&7).
// Lane's 16 halves are contiguous -> loaded as one aligned v16h in the GEMM.
// ---------------------------------------------------------------------------
__global__ void pack_weights(const float* __restrict__ W1, const float* __restrict__ U1,
                             const float* __restrict__ W2, const float* __restrict__ U2,
                             _Float16* __restrict__ out) {
  int gid = blockIdx.x * blockDim.x + threadIdx.x;
  if (gid >= 4 * 65536) return;
  int w = gid >> 16;
  int r = gid & 65535;
  const float* srcw = (w == 0) ? W1 : (w == 1) ? U1 : (w == 2) ? W2 : U2;
  int block  = r >> 9;          // 0..127
  int within = r & 511;
  int lane   = within >> 4;     // 0..31
  int h      = within & 15;     // 0..15
  int ct = block >> 2, kt = block & 3;
  int col = ct * 16 + (lane & 15);
  int kk  = kt * 32 + ((h >> 3) << 4) + ((lane >> 4) << 3) + (h & 7);
  out[gid] = (_Float16)srcw[kk * FD + col];
}

// ---------------------------------------------------------------------------
// One gated update:  s = act@W + bW + mh@U + bU  (mh/mc = segment sums over
// children via sorted dst);  i,o,u,f = chunks;  c_out = sig(i)*tanh(u)+sig(f)*mc;
// h_out = sig(o)*tanh(c_out).  One 16-node tile per workgroup, 8 waves.
// ---------------------------------------------------------------------------
__global__ __launch_bounds__(256, 2)
void stage_kernel(const float* __restrict__ act,
                  const float* __restrict__ hsrc,
                  const float* __restrict__ csrc,
                  const _Float16* __restrict__ Wp,
                  const _Float16* __restrict__ Up,
                  const float* __restrict__ bW,
                  const float* __restrict__ bU,
                  const int* __restrict__ src,
                  const int* __restrict__ dst,
                  float* __restrict__ h_out,
                  float* __restrict__ c_out,
                  int N, int E)
{
  __shared__ float lds_a [16][HD];   // activation tile (A1)
  __shared__ float lds_mh[16][HD];   // segment-summed h (A2)
  __shared__ float lds_mc[16][HD];   // segment-summed c
  __shared__ float lds_s [16][FD];   // pre-gate result
  __shared__ int   lds_beg[17];

  const int tid   = threadIdx.x;
  const int node0 = blockIdx.x * 16;

  // --- edge ranges: lower_bound over sorted dst -------------------------------
  if (tid < 17) {
    int target = node0 + tid;
    int lo = 0, hi = E;
    while (lo < hi) { int mid = (lo + hi) >> 1; if (dst[mid] < target) lo = mid + 1; else hi = mid; }
    lds_beg[tid] = lo;
  }
  __syncthreads();

  // --- gather phase: thread -> (local node, 8-float chunk) --------------------
  {
    int n = tid >> 4;
    int j = (tid & 15) * 8;
    int g = node0 + n;
    float ah[8], ac[8], ax[8];
#pragma unroll
    for (int k = 0; k < 8; ++k) { ah[k] = 0.f; ac[k] = 0.f; ax[k] = 0.f; }
    if (g < N) {
      const float* ar = act + (size_t)g * HD + j;
#pragma unroll
      for (int k = 0; k < 8; ++k) ax[k] = ar[k];
      int b = lds_beg[n], e = lds_beg[n + 1];
      for (int ed = b; ed < e; ++ed) {
        int s = src[ed];
        const float* hr = hsrc + (size_t)s * HD + j;
        const float* cr = csrc + (size_t)s * HD + j;
#pragma unroll
        for (int k = 0; k < 8; ++k) { ah[k] += hr[k]; ac[k] += cr[k]; }
      }
    }
#pragma unroll
    for (int k = 0; k < 8; ++k) {
      lds_a [n][j + k] = ax[k];
      lds_mh[n][j + k] = ah[k];
      lds_mc[n][j + k] = ac[k];
    }
  }
  __syncthreads();

  // --- WMMA phase: wave wv handles column tiles ct = wv*4 .. wv*4+3 -----------
  {
    const int lane = tid & 31;
    const int wv   = tid >> 5;
    const int m    = lane & 15;          // A row / C column lane
    const int kb   = (lane >> 4) << 3;   // K sub-base: 0 or 8

    AFrag fa[4], fm[4];                  // A fragments: act and mh, kt = 0..3
#pragma unroll
    for (int kt = 0; kt < 4; ++kt) {
#pragma unroll
      for (int h = 0; h < 16; ++h) {
        int kk = kt * 32 + ((h >> 3) << 4) + kb + (h & 7);
        fa[kt].e[h] = (_Float16)lds_a [m][kk];
        fm[kt].e[h] = (_Float16)lds_mh[m][kk];
      }
    }

    const v16h* wbase = (const v16h*)Wp;
    const v16h* ubase = (const v16h*)Up;
#pragma unroll
    for (int c = 0; c < 4; ++c) {
      int ct   = wv * 4 + c;
      int colN = ct * 16 + m;
      float bias = bW[colN] + bU[colN];  // C element depends only on N
      v8f acc = { bias, bias, bias, bias, bias, bias, bias, bias };
      const v16h* wb = wbase + (size_t)(ct * 4) * 32 + lane;
      const v16h* ub = ubase + (size_t)(ct * 4) * 32 + lane;
#pragma unroll
      for (int kt = 0; kt < 4; ++kt) {
        v16h bw = wb[kt * 32];
        acc = __builtin_amdgcn_wmma_f32_16x16x32_f16(false, fa[kt].v, false, bw,
                                                     (short)0, acc, false, false);
        v16h bu = ub[kt * 32];
        acc = __builtin_amdgcn_wmma_f32_16x16x32_f16(false, fm[kt].v, false, bu,
                                                     (short)0, acc, false, false);
      }
      CFrag cf; cf.v = acc;
      int mrow = (lane >> 4) << 3;       // M base: 0 or 8
#pragma unroll
      for (int r = 0; r < 8; ++r) lds_s[mrow + r][colN] = cf.e[r];
    }
  }
  __syncthreads();

  // --- gate phase (chunk order: i, o, u, f) -----------------------------------
  {
    int n = tid >> 4;
    int j = (tid & 15) * 8;
    int g = node0 + n;
    if (g < N) {
#pragma unroll
      for (int k = 0; k < 8; ++k) {
        int c = j + k;
        float i = sigf (lds_s[n][c]);
        float o = sigf (lds_s[n][HD + c]);
        float u = tanhf(lds_s[n][2 * HD + c]);
        float f = sigf (lds_s[n][3 * HD + c]);
        float cn = i * u + f * lds_mc[n][c];
        float hn = o * tanhf(cn);
        c_out[(size_t)g * HD + c] = cn;
        h_out[(size_t)g * HD + c] = hn;
      }
    }
  }
}

extern "C" void kernel_launch(void* const* d_in, const int* in_sizes, int n_in,
                              void* d_out, int out_size, void* d_ws, size_t ws_size,
                              hipStream_t stream) {
  const float* x   = (const float*)d_in[0];
  const float* h1  = (const float*)d_in[1];
  const float* c1  = (const float*)d_in[2];
  const float* h2  = (const float*)d_in[3];
  const float* c2  = (const float*)d_in[4];
  const float* W1  = (const float*)d_in[5];
  const float* bW1 = (const float*)d_in[6];
  const float* U1  = (const float*)d_in[7];
  const float* bU1 = (const float*)d_in[8];
  const float* W2  = (const float*)d_in[9];
  const float* bW2 = (const float*)d_in[10];
  const float* U2  = (const float*)d_in[11];
  const float* bU2 = (const float*)d_in[12];
  const int*   src = (const int*)d_in[13];
  const int*   dst = (const int*)d_in[14];

  const int N = in_sizes[0] / HD;
  const int E = in_sizes[13];

  _Float16* Wp1 = (_Float16*)d_ws;      // 4 x 64K halves = 512 KB total
  _Float16* Up1 = Wp1 + 65536;
  _Float16* Wp2 = Up1 + 65536;
  _Float16* Up2 = Wp2 + 65536;

  float* out = (float*)d_out;           // [h1n | c1n | h2n | c2n]
  float* h1n = out;
  float* c1n = out + 1 * (size_t)N * HD;
  float* h2n = out + 2 * (size_t)N * HD;
  float* c2n = out + 3 * (size_t)N * HD;

  pack_weights<<<(4 * 65536 + 255) / 256, 256, 0, stream>>>(W1, U1, W2, U2, Wp1);

  int tiles = (N + 15) / 16;
  stage_kernel<<<tiles, 256, 0, stream>>>(x,   h1, c1, Wp1, Up1, bW1, bU1, src, dst, h1n, c1n, N, E);
  stage_kernel<<<tiles, 256, 0, stream>>>(c1n, h2, c2, Wp2, Up2, bW2, bU2, src, dst, h2n, c2n, N, E);
}
